// GAT_9878424781129
// MI455X (gfx1250) — matrix operand order repaired and checked by reading, not compiled, not run on hardware
//
#include <hip/hip_runtime.h>

typedef __bf16 v16bf __attribute__((ext_vector_type(16)));
typedef __bf16 v8bf  __attribute__((ext_vector_type(8)));
typedef float  v8f   __attribute__((ext_vector_type(8)));

// ---------- helpers ----------
__device__ inline float wave_sum(float v) {
#pragma unroll
    for (int o = 16; o; o >>= 1) v += __shfl_xor(v, o, 32);
    return v;
}
// monotonic uint encoding of float for atomicMax (handles negatives)
__device__ inline unsigned fkey(float f) {
    unsigned b = __float_as_uint(f);
    return (b & 0x80000000u) ? ~b : (b | 0x80000000u);
}
__device__ inline float fdecode(unsigned k) {
    unsigned b = (k & 0x80000000u) ? (k & 0x7fffffffu) : ~k;
    return __uint_as_float(b);
}
// low 32 bits of a generic pointer to LDS == AS3/LDS byte address (ISA flat aperture rule)
__device__ __forceinline__ unsigned lds_off(const void* p) {
    return (unsigned)(unsigned long long)(uintptr_t)p;
}
// CDNA5 async copy: 16B per lane, global -> LDS, tracked by ASYNCcnt
__device__ __forceinline__ void async_ld16(unsigned ldsoff, const void* gptr) {
    asm volatile("global_load_async_to_lds_b128 %0, %1, off"
                 :: "v"(ldsoff), "v"(gptr) : "memory");
}
__device__ __forceinline__ v16bf cat16(v8bf lo, v8bf hi) {
    return __builtin_shufflevector(lo, hi, 0,1,2,3,4,5,6,7,8,9,10,11,12,13,14,15);
}

// ---------- conversion ----------
__global__ void gat_f32_to_bf16(const float* __restrict__ in, __bf16* __restrict__ out, size_t n) {
    size_t t = (size_t)blockIdx.x * blockDim.x + threadIdx.x;
    if (t < n) out[t] = (__bf16)in[t];
}

// ---------- WMMA GEMM with async-LDS double buffering ----------
// C[M,N] = A[M,K] * B[N,K]^T ; A,B bf16 row-major; C f32.
// Block 256 threads = 8 waves (4 M x 2 N); tile 128(M) x 64(N); K stage 64.
#define MTILE 128
#define NTILE 64
#define KTILE 64
#define LPAD  72   // LDS row stride in elements (144B, breaks bank conflicts, 16B aligned)

__global__ __launch_bounds__(256) void gat_wmma_gemm_lds(
    const __bf16* __restrict__ A, const __bf16* __restrict__ B, float* __restrict__ C,
    int M, int N, int K) {
    __shared__ __bf16 sA[2][MTILE * LPAD];
    __shared__ __bf16 sB[2][NTILE * LPAD];

    const int NB = N / NTILE;
    const int bm = blockIdx.x / NB, bn = blockIdx.x % NB;
    const int m0 = bm * MTILE, n0 = bn * NTILE;
    const int tid  = threadIdx.x;
    const int wave = tid >> 5, lane = tid & 31;
    const int wm = wave >> 1, wn = wave & 1;      // 4 x 2 wave grid
    const int half = lane >> 4, l16 = lane & 15;

    auto issue_stage = [&](int k0, int b) {
        // A tile: 128 rows x 64 elems = 1024 x 16B chunks; 4 per thread
#pragma unroll
        for (int t = 0; t < 4; ++t) {
            int c = tid + t * 256;
            int row = c >> 3, kc = c & 7;
            int gr = m0 + row; gr = (gr < M) ? gr : (M - 1);   // clamp M tail
            async_ld16(lds_off(&sA[b][row * LPAD + kc * 8]),
                       A + (size_t)gr * K + k0 + kc * 8);
        }
        // B tile: 64 rows x 64 elems = 512 x 16B chunks; 2 per thread
#pragma unroll
        for (int t = 0; t < 2; ++t) {
            int c = tid + t * 256;
            int row = c >> 3, kc = c & 7;
            async_ld16(lds_off(&sB[b][row * LPAD + kc * 8]),
                       B + (size_t)(n0 + row) * K + k0 + kc * 8);
        }
    };

    v8f acc[2][2] = {};

    const int S = K / KTILE;
    issue_stage(0, 0);
    for (int s = 0; s < S; ++s) {
        if (s + 1 < S) {
            issue_stage((s + 1) * KTILE, (s + 1) & 1);
            asm volatile("s_wait_asynccnt 0x6" ::: "memory");   // stage s landed (in-order)
        } else {
            asm volatile("s_wait_asynccnt 0x0" ::: "memory");
        }
        __syncthreads();

        const int b = s & 1;
#pragma unroll
        for (int ks = 0; ks < 2; ++ks) {
            v16bf af[2], bf[2];
#pragma unroll
            for (int mi = 0; mi < 2; ++mi) {
                // A frag: lane half gives K {+0..7,+16..23} / {+8..15,+24..31}
                const __bf16* p = &sA[b][(wm * 32 + mi * 16 + l16) * LPAD + ks * 32 + half * 8];
                af[mi] = cat16(*(const v8bf*)p, *(const v8bf*)(p + 16));
            }
#pragma unroll
            for (int ni = 0; ni < 2; ++ni) {
                // B frag: lane l16 = column; lane half gives K {+0..7,+16..23} / {+8..15,+24..31}
                const __bf16* p = &sB[b][(wn * 32 + ni * 16 + l16) * LPAD + ks * 32 + half * 8];
                bf[ni] = cat16(*(const v8bf*)p, *(const v8bf*)(p + 16));
            }
#pragma unroll
            for (int mi = 0; mi < 2; ++mi)
#pragma unroll
                for (int ni = 0; ni < 2; ++ni)
                    acc[mi][ni] = __builtin_amdgcn_wmma_f32_16x16x32_bf16(
                        false, af[mi], false, bf[ni], (short)0, acc[mi][ni], false, false);
        }
        __syncthreads();   // done reading buf b before it is overwritten
    }

    // C/D layout: lane half selects rows 0..7 / 8..15; l16 = column
#pragma unroll
    for (int mi = 0; mi < 2; ++mi)
#pragma unroll
        for (int ni = 0; ni < 2; ++ni) {
            int col = n0 + wn * 32 + ni * 16 + l16;
            int rbase = m0 + wm * 32 + mi * 16 + half * 8;
#pragma unroll
            for (int r = 0; r < 8; ++r) {
                int row = rbase + r;
                if (row < M) C[(size_t)row * N + col] = acc[mi][ni][r];
            }
        }
}

// ---------- per-(node,head) attention coefficients: one wave per row ----------
__global__ void gat_alpha(const float* __restrict__ h, const float* __restrict__ a_src,
                          const float* __restrict__ a_dst, float* __restrict__ os,
                          float* __restrict__ od, int rows, int C, int H) {
    int wave = blockIdx.x * (blockDim.x >> 5) + (threadIdx.x >> 5);
    if (wave >= rows) return;
    int lane = threadIdx.x & 31;
    int hd = wave % H;
    const float* hp = h + (size_t)wave * C;
    const float* as = a_src + (size_t)hd * C;
    const float* ad = a_dst + (size_t)hd * C;
    float ss = 0.f, sd = 0.f;
    for (int c = lane; c < C; c += 32) { float v = hp[c]; ss += v * as[c]; sd += v * ad[c]; }
    ss = wave_sum(ss); sd = wave_sum(sd);
    if (lane == 0) { os[wave] = ss; od[wave] = sd; }
}

// ---------- edge pass 1: leaky-relu score + segment max ----------
__global__ void gat_edge1(const float* __restrict__ asrc, const float* __restrict__ adst,
                          const int* __restrict__ esrc, const int* __restrict__ edst,
                          float* __restrict__ eedge, unsigned* __restrict__ segmax,
                          int E, int ET, int H) {
    int t = blockIdx.x * blockDim.x + threadIdx.x;
    if (t >= ET * H) return;
    int i = t / H, hd = t - i * H;
    int s = (i < E) ? esrc[i] : (i - E);
    int d = (i < E) ? edst[i] : (i - E);
    float v = asrc[s * H + hd] + adst[d * H + hd];
    v = (v > 0.f) ? v : 0.2f * v;
    eedge[t] = v;
    atomicMax(&segmax[d * H + hd], fkey(v));
}

// ---------- edge pass 2: exp + segment sum ----------
__global__ void gat_edge2(const float* __restrict__ eedge, const int* __restrict__ edst,
                          const unsigned* __restrict__ segmax, float* __restrict__ exedge,
                          float* __restrict__ segsum, int E, int ET, int H) {
    int t = blockIdx.x * blockDim.x + threadIdx.x;
    if (t >= ET * H) return;
    int i = t / H, hd = t - i * H;
    int d = (i < E) ? edst[i] : (i - E);
    float m = fdecode(segmax[d * H + hd]);
    if (!(m > -1e38f && m < 1e38f)) m = 0.f;     // matches where(isfinite(m), m, 0)
    float ex = __expf(eedge[t] - m);
    exedge[t] = ex;
    atomicAdd(&segsum[d * H + hd], ex);
}

// ---------- edge pass 3: weighted aggregation (one block per edge) ----------
__global__ void gat_edge_agg(const float* __restrict__ hmat, const float* __restrict__ exedge,
                             const float* __restrict__ segsum, const int* __restrict__ esrc,
                             const int* __restrict__ edst, float* __restrict__ out,
                             int E, int H, int C) {
    int i = blockIdx.x;
    int s = (i < E) ? esrc[i] : (i - E);
    int d = (i < E) ? edst[i] : (i - E);
    int HC = H * C;
    const float* hs = hmat + (size_t)s * HC;
    float* po = out + (size_t)d * HC;
    for (int ch = threadIdx.x; ch < HC; ch += blockDim.x) {
        int hd = ch / C;
        float coef = exedge[(size_t)i * H + hd] / (segsum[d * H + hd] + 1e-16f);
        atomicAdd(&po[ch], hs[ch] * coef);
    }
}

// ---------- bias + ELU; optional f32 and bf16 outputs ----------
__global__ void gat_finalize(const float* __restrict__ agg, const float* __restrict__ bias,
                             float* __restrict__ of, __bf16* __restrict__ ob,
                             size_t total, int HC) {
    size_t t = (size_t)blockIdx.x * blockDim.x + threadIdx.x;
    if (t >= total) return;
    float v = agg[t] + bias[t % HC];
    v = (v > 0.f) ? v : (__expf(v) - 1.f);
    if (of) of[t] = v;
    if (ob) ob[t] = (__bf16)v;
}

// ---------- classifier ----------
__global__ void gat_logits(const float* __restrict__ z, const float* __restrict__ clsW,
                           const float* __restrict__ clsb, float* __restrict__ out, int N) {
    int t = blockIdx.x * blockDim.x + threadIdx.x;
    if (t >= N * 4) return;
    int n = t >> 2, j = t & 3;
    const float* zp = z + (size_t)n * 128;
    const float* wp = clsW + (size_t)j * 128;
    float s = 0.f;
#pragma unroll 8
    for (int k = 0; k < 128; ++k) s += zp[k] * wp[k];
    out[t] = s + clsb[j];
}

// ---------- link predictions: one wave per query edge ----------
__global__ void gat_pred(const float* __restrict__ z, const int* __restrict__ p0,
                         const int* __restrict__ p1, const int* __restrict__ n0,
                         const int* __restrict__ n1, float* __restrict__ out, int P) {
    int wave = blockIdx.x * (blockDim.x >> 5) + (threadIdx.x >> 5);
    if (wave >= 2 * P) return;
    int lane = threadIdx.x & 31;
    int id; const int *ea, *eb;
    if (wave < P) { ea = p0; eb = p1; id = wave; } else { ea = n0; eb = n1; id = wave - P; }
    int u = ea[id], v = eb[id];
    const float* zu = z + (size_t)u * 128;
    const float* zv = z + (size_t)v * 128;
    float s = 0.f;
    for (int c = lane; c < 128; c += 32) s += zu[c] * zv[c];
    s = wave_sum(s);
    if (lane == 0) out[wave] = 1.f / (1.f + __expf(-s));
}

extern "C" void kernel_launch(void* const* d_in, const int* in_sizes, int n_in,
                              void* d_out, int out_size, void* d_ws, size_t ws_size,
                              hipStream_t stream) {
    const float* x      = (const float*)d_in[0];
    const int*   e      = (const int*)d_in[1];
    const int*   pe     = (const int*)d_in[2];
    const int*   ne     = (const int*)d_in[3];
    const float* W1     = (const float*)d_in[4];
    const float* a_src1 = (const float*)d_in[5];
    const float* a_dst1 = (const float*)d_in[6];
    const float* b1     = (const float*)d_in[7];
    const float* W2     = (const float*)d_in[8];
    const float* a_src2 = (const float*)d_in[9];
    const float* a_dst2 = (const float*)d_in[10];
    const float* b2     = (const float*)d_in[11];
    const float* clsW   = (const float*)d_in[12];
    const float* clsb   = (const float*)d_in[13];

    constexpr int N = 50000, DIM = 128, H1 = 4, HID1 = 256, HC1 = 1024, EMB = 128;
    constexpr int E = 200000, ET = 250000, P = 50000;

    char* w = (char*)d_ws;
    auto take = [&](size_t bytes) { char* r = w; w += (bytes + 255) & ~(size_t)255; return r; };
    __bf16*   xb     = (__bf16*)take((size_t)N * DIM * 2);
    __bf16*   W1b    = (__bf16*)take((size_t)HC1 * DIM * 2);
    __bf16*   W2b    = (__bf16*)take((size_t)EMB * HC1 * 2);
    float*    h      = (float*)take((size_t)N * HC1 * 4);   // GEMM out (layer1) / reused layer2
    float*    agg    = (float*)take((size_t)N * HC1 * 4);   // aggregation (layer1) / reused layer2
    __bf16*   h1b    = (__bf16*)take((size_t)N * HC1 * 2);  // bf16 h1 for GEMM2
    float*    asrc   = (float*)take((size_t)N * H1 * 4);
    float*    adst   = (float*)take((size_t)N * H1 * 4);
    unsigned* segmax = (unsigned*)take((size_t)N * H1 * 4);
    float*    segsum = (float*)take((size_t)N * H1 * 4);
    float*    eedge  = (float*)take((size_t)ET * H1 * 4);
    float*    exedge = (float*)take((size_t)ET * H1 * 4);
    float*    h2   = h;    // safe reuse after layer-1 aggregation done
    float*    agg2 = agg;  // safe reuse after layer-1 finalize done

    float* z      = (float*)d_out;                  // [N,128]
    float* logits = z + (size_t)N * EMB;            // [N,4]
    float* preds  = logits + (size_t)N * 4;         // [2P]

    const int* esrc = e;
    const int* edst = e + E;

    // bf16 conversions
    gat_f32_to_bf16<<<(N * DIM + 255) / 256, 256, 0, stream>>>(x, xb, (size_t)N * DIM);
    gat_f32_to_bf16<<<(HC1 * DIM + 255) / 256, 256, 0, stream>>>(W1, W1b, (size_t)HC1 * DIM);
    gat_f32_to_bf16<<<(EMB * HC1 + 255) / 256, 256, 0, stream>>>(W2, W2b, (size_t)EMB * HC1);

    const int MB = (N + MTILE - 1) / MTILE;  // 391

    // ---------------- layer 1 ----------------
    gat_wmma_gemm_lds<<<MB * (HC1 / NTILE), 256, 0, stream>>>(xb, W1b, h, N, HC1, DIM);
    {
        int rows = N * H1;
        gat_alpha<<<(rows + 7) / 8, 256, 0, stream>>>(h, a_src1, a_dst1, asrc, adst, rows, HID1, H1);
    }
    hipMemsetAsync(segmax, 0, (size_t)N * H1 * 4, stream);  // key 0 decodes to NaN -> m=0 path
    hipMemsetAsync(segsum, 0, (size_t)N * H1 * 4, stream);
    hipMemsetAsync(agg, 0, (size_t)N * HC1 * 4, stream);
    gat_edge1<<<(ET * H1 + 255) / 256, 256, 0, stream>>>(asrc, adst, esrc, edst, eedge, segmax, E, ET, H1);
    gat_edge2<<<(ET * H1 + 255) / 256, 256, 0, stream>>>(eedge, edst, segmax, exedge, segsum, E, ET, H1);
    gat_edge_agg<<<ET, 256, 0, stream>>>(h, exedge, segsum, esrc, edst, agg, E, H1, HID1);
    gat_finalize<<<(unsigned)(((size_t)N * HC1 + 255) / 256), 256, 0, stream>>>(
        agg, b1, nullptr, h1b, (size_t)N * HC1, HC1);

    // ---------------- layer 2 ----------------
    gat_wmma_gemm_lds<<<MB * (EMB / NTILE), 256, 0, stream>>>(h1b, W2b, h2, N, EMB, HC1);
    gat_alpha<<<(N + 7) / 8, 256, 0, stream>>>(h2, a_src2, a_dst2, asrc, adst, N, EMB, 1);
    hipMemsetAsync(segmax, 0, (size_t)N * 4, stream);
    hipMemsetAsync(segsum, 0, (size_t)N * 4, stream);
    hipMemsetAsync(agg2, 0, (size_t)N * EMB * 4, stream);
    gat_edge1<<<(ET + 255) / 256, 256, 0, stream>>>(asrc, adst, esrc, edst, eedge, segmax, E, ET, 1);
    gat_edge2<<<(ET + 255) / 256, 256, 0, stream>>>(eedge, edst, segmax, exedge, segsum, E, ET, 1);
    gat_edge_agg<<<ET, 128, 0, stream>>>(h2, exedge, segsum, esrc, edst, agg2, E, 1, EMB);
    gat_finalize<<<(unsigned)(((size_t)N * EMB + 255) / 256), 256, 0, stream>>>(
        agg2, b2, z, nullptr, (size_t)N * EMB, EMB);

    // ---------------- heads ----------------
    gat_logits<<<(N * 4 + 255) / 256, 256, 0, stream>>>(z, clsW, clsb, logits, N);
    gat_pred<<<(2 * P + 7) / 8, 256, 0, stream>>>(z, pe, pe + P, ne, ne + P, preds, P);
}
